// MoELayer_52888227283710
// MI455X (gfx1250) — compile-verified, matrix operand order checked by reading, not run on hardware
//
#include <hip/hip_runtime.h>
#include <hip/hip_bf16.h>
#include <math.h>

// ---------------- problem constants ----------------
#define D_MODEL 1024
#define D_FFN   2048
#define N_EXP   8
#define N_TOK   4096            // B*S = 2*2048
#define SLOTS   (N_TOK*2)       // top-2 assignments
#define LDT     40              // LDS row stride (bf16 elems), 80B -> keeps 16B align

// ---------------- workspace layout (bytes). total ~72.3 MB ----------------
#define OFF_COUNTS 0u                        // 8 x int
#define OFF_CNT2   64u                       // 8 x int
#define OFF_BASE   128u                      // 8 x int
#define OFF_EIDX   256u                      // 2*N_TOK int   (32 KB)
#define OFF_WTOK   (OFF_EIDX + 8u*N_TOK)     // 2*N_TOK float (32 KB)
#define OFF_TOK    (OFF_WTOK + 8u*N_TOK)     // SLOTS int
#define OFF_WGT    (OFF_TOK  + 4u*SLOTS)     // SLOTS float
#define OFF_INV    (OFF_WGT  + 4u*SLOTS)     // 2*N_TOK int
#define OFF_XB     262144u                               // bf16 x  : 8 MB
#define OFF_H      (OFF_XB + 2ull*N_TOK*D_MODEL)         // bf16 H  : 32 MB
#define OFF_Y      (OFF_H  + 2ull*SLOTS*D_FFN)           // f32  Y  : 32 MB

typedef __bf16 bf16x16 __attribute__((ext_vector_type(16)));
typedef float  f32x8   __attribute__((ext_vector_type(8)));
typedef int    v4i     __attribute__((ext_vector_type(4)));
union FragU { unsigned u[8]; bf16x16 v; };

// ---- async global->LDS (CDNA5 GLOBAL_LOAD_ASYNC_TO_LDS_B128, ASYNCcnt) ----
#if defined(__gfx1250__) && __has_builtin(__builtin_amdgcn_global_load_async_to_lds_b128)
#define HAVE_ASYNC_LDS 1
#else
#define HAVE_ASYNC_LDS 0
#endif

__device__ __forceinline__ void async_copy16(unsigned short* lds_dst,
                                             const unsigned short* gsrc) {
#if HAVE_ASYNC_LDS
  __builtin_amdgcn_global_load_async_to_lds_b128(
      (__attribute__((address_space(1))) v4i*)(gsrc),
      (__attribute__((address_space(3))) v4i*)(lds_dst),
      0, 0);
#else
  *(uint4*)lds_dst = *(const uint4*)gsrc;
#endif
}

__device__ __forceinline__ void wait_async() {
#if HAVE_ASYNC_LDS
# if __has_builtin(__builtin_amdgcn_s_wait_asynccnt)
  __builtin_amdgcn_s_wait_asynccnt(0);
# else
  asm volatile("s_wait_asynccnt 0" ::: "memory");
# endif
#endif
}

__device__ __forceinline__ unsigned short f32_bf16(float f) {
  unsigned u = __float_as_uint(f);
  u += 0x7FFFu + ((u >> 16) & 1u);          // round-to-nearest-even
  return (unsigned short)(u >> 16);
}
__device__ __forceinline__ unsigned pack2(float a, float b) {
  return (unsigned)f32_bf16(a) | ((unsigned)f32_bf16(b) << 16);
}

// Gather one 16x32 bf16 A/B fragment from LDS per the CDNA5 16-bit layout:
// lanes 0-15: row=r, K in {0..7, 16..23}; lanes 16-31: row=r, K in {8..15, 24..31}
__device__ __forceinline__ void load_frag(const unsigned short* s, int row0, FragU& f) {
  int lane = threadIdx.x & 31;
  const unsigned short* p = s + (row0 + (lane & 15)) * LDT + ((lane >> 4) << 3);
  uint4 a = *(const uint4*)p;          // K = half*8 .. +7   (ds_load_b128)
  uint4 b = *(const uint4*)(p + 16);   // K = 16+half*8 .. +7
  f.u[0]=a.x; f.u[1]=a.y; f.u[2]=a.z; f.u[3]=a.w;
  f.u[4]=b.x; f.u[5]=b.y; f.u[6]=b.z; f.u[7]=b.w;
}

__device__ __forceinline__ f32x8 wmma_bf16(const FragU& a, const FragU& b, f32x8 c) {
  return __builtin_amdgcn_wmma_f32_16x16x32_bf16(false, a.v, false, b.v,
                                                 (short)0, c, false, false);
}

// ---------------- small kernels ----------------
__global__ void init_kernel(int* counts, int* cnt2) {
  int t = threadIdx.x;
  if (t < N_EXP) { counts[t] = 0; cnt2[t] = 0; }
}

__global__ void xb_kernel(const float* __restrict__ x, unsigned short* __restrict__ xb) {
  int i = (blockIdx.x * 256 + threadIdx.x) * 4;
  float4 f = *(const float4*)(x + i);
  uint2 o; o.x = pack2(f.x, f.y); o.y = pack2(f.z, f.w);
  *(uint2*)(xb + i) = o;
}

// one wave32 per token: fp32 logits, top-2, softmax over the 2
__global__ void router_kernel(const float* __restrict__ x, const float* __restrict__ wr,
                              int* __restrict__ eIdx, float* __restrict__ wTok,
                              int* counts) {
  int wid = threadIdx.x >> 5, lane = threadIdx.x & 31;
  int t = blockIdx.x * 8 + wid;
  float s[8];
#pragma unroll
  for (int e = 0; e < 8; e++) s[e] = 0.f;
  const float* xp = x + (size_t)t * D_MODEL;
  for (int k = lane; k < D_MODEL; k += 32) {
    float xv = xp[k];
#pragma unroll
    for (int e = 0; e < 8; e++) s[e] += xv * wr[e * D_MODEL + k];
  }
#pragma unroll
  for (int off = 16; off; off >>= 1) {
#pragma unroll
    for (int e = 0; e < 8; e++) s[e] += __shfl_xor(s[e], off);
  }
  if (lane == 0) {
    int i1 = 0; float v1 = s[0];
    for (int e = 1; e < 8; e++) if (s[e] > v1) { v1 = s[e]; i1 = e; }
    int i2 = -1; float v2 = -3.4e38f;
    for (int e = 0; e < 8; e++) if (e != i1 && s[e] > v2) { v2 = s[e]; i2 = e; }
    float ex = __expf(v2 - v1);
    float inv = 1.f / (1.f + ex);
    eIdx[2*t] = i1; eIdx[2*t+1] = i2;
    wTok[2*t] = inv; wTok[2*t+1] = ex * inv;
    atomicAdd(&counts[i1], 1); atomicAdd(&counts[i2], 1);
  }
}

__global__ void scan_kernel(const int* counts, int* base) {
  if (threadIdx.x == 0) {
    int a = 0;
    for (int e = 0; e < N_EXP; e++) { base[e] = a; a += counts[e]; }
  }
}

__global__ void scatter_kernel(const int* __restrict__ eIdx, const float* __restrict__ wTok,
                               const int* __restrict__ base, int* cnt2,
                               int* __restrict__ tok, float* __restrict__ wgt,
                               int* __restrict__ inv) {
  int t = blockIdx.x * 256 + threadIdx.x;
  if (t >= N_TOK) return;
#pragma unroll
  for (int k = 0; k < 2; k++) {
    int e = eIdx[2*t + k];
    int p = base[e] + atomicAdd(&cnt2[e], 1);
    tok[p] = t; wgt[p] = wTok[2*t + k]; inv[2*t + k] = p;
  }
}

// ---------------- expert GEMM 1: H = silu(X Wg^T) * (X Wu^T) ----------------
// block 256 thr (8 waves), tile BM=128 x BN=64, BK=32; wave = 32x32 C tile x2 mats
__global__ void gemm_gateup_kernel(const unsigned short* __restrict__ xb,
                                   const float* __restrict__ w_gate,
                                   const float* __restrict__ w_up,
                                   unsigned short* __restrict__ H,
                                   const int* __restrict__ counts,
                                   const int* __restrict__ base,
                                   const int* __restrict__ tok) {
  int e  = blockIdx.z;
  int nE = counts[e];
  int m0 = blockIdx.x * 128;
  if (m0 >= nE) return;
  int n0 = blockIdx.y * 64;
  int b0 = base[e];
  int tid = threadIdx.x;

  __shared__ __align__(16) unsigned short Xs[128 * LDT];
  __shared__ __align__(16) unsigned short Gs[64 * LDT];
  __shared__ __align__(16) unsigned short Us[64 * LDT];

  int lr  = tid >> 2;              // 0..63
  int seg = tid & 3;               // k sub-offset seg*8
  int tokA = (m0 + lr      < nE) ? tok[b0 + m0 + lr]      : -1;
  int tokB = (m0 + lr + 64 < nE) ? tok[b0 + m0 + lr + 64] : -1;
  const float* wgp = w_gate + ((size_t)e * D_FFN + (n0 + lr)) * D_MODEL + seg * 8;
  const float* wup = w_up   + ((size_t)e * D_FFN + (n0 + lr)) * D_MODEL + seg * 8;

  // invalid rows are never written in the k-loop: zero them exactly once
  uint4 z4 = {0,0,0,0};
  if (tokA < 0) *(uint4*)(&Xs[lr * LDT + seg * 8]) = z4;
  if (tokB < 0) *(uint4*)(&Xs[(lr + 64) * LDT + seg * 8]) = z4;

  int w = tid >> 5, wm = w & 3, wn = w >> 2;

  f32x8 zero = {0.f,0.f,0.f,0.f,0.f,0.f,0.f,0.f};
  f32x8 aG[2][2], aU[2][2];
#pragma unroll
  for (int i = 0; i < 2; i++)
#pragma unroll
    for (int j = 0; j < 2; j++) { aG[i][j] = zero; aU[i][j] = zero; }

  for (int k0 = 0; k0 < D_MODEL; k0 += 32) {
    // X tile: pure bf16 copy -> async global->LDS path
    if (tokA >= 0)
      async_copy16(&Xs[lr * LDT + seg * 8],
                   xb + (size_t)tokA * D_MODEL + k0 + seg * 8);
    if (tokB >= 0)
      async_copy16(&Xs[(lr + 64) * LDT + seg * 8],
                   xb + (size_t)tokB * D_MODEL + k0 + seg * 8);
    // weight tiles: fp32 -> bf16 conversion must go through VALU
    float4 g0 = *(const float4*)(wgp + k0);
    float4 g1 = *(const float4*)(wgp + k0 + 4);
    float4 u0 = *(const float4*)(wup + k0);
    float4 u1 = *(const float4*)(wup + k0 + 4);
    uint4 gg; gg.x = pack2(g0.x,g0.y); gg.y = pack2(g0.z,g0.w);
              gg.z = pack2(g1.x,g1.y); gg.w = pack2(g1.z,g1.w);
    *(uint4*)(&Gs[lr * LDT + seg * 8]) = gg;
    uint4 uu; uu.x = pack2(u0.x,u0.y); uu.y = pack2(u0.z,u0.w);
              uu.z = pack2(u1.x,u1.y); uu.w = pack2(u1.z,u1.w);
    *(uint4*)(&Us[lr * LDT + seg * 8]) = uu;
    wait_async();
    __syncthreads();

    FragU a0, a1, bg0, bg1, bu0, bu1;
    load_frag(Xs, wm * 32,      a0);
    load_frag(Xs, wm * 32 + 16, a1);
    load_frag(Gs, wn * 32,      bg0);
    load_frag(Gs, wn * 32 + 16, bg1);
    load_frag(Us, wn * 32,      bu0);
    load_frag(Us, wn * 32 + 16, bu1);
    aG[0][0] = wmma_bf16(a0, bg0, aG[0][0]); aG[0][1] = wmma_bf16(a0, bg1, aG[0][1]);
    aG[1][0] = wmma_bf16(a1, bg0, aG[1][0]); aG[1][1] = wmma_bf16(a1, bg1, aG[1][1]);
    aU[0][0] = wmma_bf16(a0, bu0, aU[0][0]); aU[0][1] = wmma_bf16(a0, bu1, aU[0][1]);
    aU[1][0] = wmma_bf16(a1, bu0, aU[1][0]); aU[1][1] = wmma_bf16(a1, bu1, aU[1][1]);
    __syncthreads();
  }

  // fused SwiGLU epilogue; C layout: VGPR r -> row r (+8 for lanes 16..31), col = lane%16
  int lane = tid & 31, colL = lane & 15, rsel = lane >> 4;
#pragma unroll
  for (int i = 0; i < 2; i++)
#pragma unroll
    for (int j = 0; j < 2; j++) {
      int col  = n0 + wn * 32 + j * 16 + colL;
      int rowB = m0 + wm * 32 + i * 16 + rsel * 8;
#pragma unroll
      for (int r = 0; r < 8; r++) {
        int mr = rowB + r;
        if (mr < nE) {
          float g = aG[i][j][r], u = aU[i][j][r];
          float h = g / (1.f + __expf(-g)) * u;     // silu(g)*u
          H[(size_t)(b0 + mr) * D_FFN + col] = f32_bf16(h);
        }
      }
    }
}

// ---------------- expert GEMM 2: Y = H Wd^T ----------------
__global__ void gemm_down_kernel(const unsigned short* __restrict__ H,
                                 const float* __restrict__ w_down,
                                 float* __restrict__ Y,
                                 const int* __restrict__ counts,
                                 const int* __restrict__ base) {
  int e  = blockIdx.z;
  int nE = counts[e];
  int m0 = blockIdx.x * 128;
  if (m0 >= nE) return;
  int n0 = blockIdx.y * 64;
  int b0 = base[e];
  int tid = threadIdx.x;

  __shared__ __align__(16) unsigned short Hs[128 * LDT];
  __shared__ __align__(16) unsigned short Ws[64 * LDT];

  int lr = tid >> 2, seg = tid & 3;
  bool vA = (m0 + lr < nE), vB = (m0 + lr + 64 < nE);
  const unsigned short* hA = H + (size_t)(b0 + m0 + lr) * D_FFN + seg * 8;
  const unsigned short* hB = H + (size_t)(b0 + m0 + lr + 64) * D_FFN + seg * 8;
  const float* wdp = w_down + ((size_t)e * D_MODEL + (n0 + lr)) * D_FFN + seg * 8;

  uint4 z4 = {0,0,0,0};
  if (!vA) *(uint4*)(&Hs[lr * LDT + seg * 8]) = z4;
  if (!vB) *(uint4*)(&Hs[(lr + 64) * LDT + seg * 8]) = z4;

  int w = tid >> 5, wm = w & 3, wn = w >> 2;
  f32x8 zero = {0.f,0.f,0.f,0.f,0.f,0.f,0.f,0.f};
  f32x8 acc[2][2];
#pragma unroll
  for (int i = 0; i < 2; i++)
#pragma unroll
    for (int j = 0; j < 2; j++) acc[i][j] = zero;

  for (int k0 = 0; k0 < D_FFN; k0 += 32) {
    // H tile: pure bf16 copy -> async global->LDS path
    if (vA) async_copy16(&Hs[lr * LDT + seg * 8], hA + k0);
    if (vB) async_copy16(&Hs[(lr + 64) * LDT + seg * 8], hB + k0);
    float4 d0 = *(const float4*)(wdp + k0);
    float4 d1 = *(const float4*)(wdp + k0 + 4);
    uint4 dd; dd.x = pack2(d0.x,d0.y); dd.y = pack2(d0.z,d0.w);
              dd.z = pack2(d1.x,d1.y); dd.w = pack2(d1.z,d1.w);
    *(uint4*)(&Ws[lr * LDT + seg * 8]) = dd;
    wait_async();
    __syncthreads();

    FragU a0, a1, c0, c1;
    load_frag(Hs, wm * 32,      a0);
    load_frag(Hs, wm * 32 + 16, a1);
    load_frag(Ws, wn * 32,      c0);
    load_frag(Ws, wn * 32 + 16, c1);
    acc[0][0] = wmma_bf16(a0, c0, acc[0][0]); acc[0][1] = wmma_bf16(a0, c1, acc[0][1]);
    acc[1][0] = wmma_bf16(a1, c0, acc[1][0]); acc[1][1] = wmma_bf16(a1, c1, acc[1][1]);
    __syncthreads();
  }

  int lane = tid & 31, colL = lane & 15, rsel = lane >> 4;
#pragma unroll
  for (int i = 0; i < 2; i++)
#pragma unroll
    for (int j = 0; j < 2; j++) {
      int col  = n0 + wn * 32 + j * 16 + colL;
      int rowB = m0 + wm * 32 + i * 16 + rsel * 8;
#pragma unroll
      for (int r = 0; r < 8; r++) {
        int mr = rowB + r;
        if (mr < nE) Y[(size_t)(b0 + mr) * D_MODEL + col] = acc[i][j][r];
      }
    }
}

// ---------------- combine top-2 + LayerNorm (one block per token) ----------------
__global__ void combine_ln_kernel(const float* __restrict__ Y, const int* __restrict__ inv,
                                  const float* __restrict__ wgt,
                                  const float* __restrict__ gamma, const float* __restrict__ beta,
                                  float* __restrict__ out) {
  int t = blockIdx.x, tid = threadIdx.x;
  int s0 = inv[2*t], s1 = inv[2*t + 1];
  float w0 = wgt[s0], w1 = wgt[s1];
  float v[4]; float lsum = 0.f;
#pragma unroll
  for (int i = 0; i < 4; i++) {
    int c = tid + i * 256;
    v[i] = w0 * Y[(size_t)s0 * D_MODEL + c] + w1 * Y[(size_t)s1 * D_MODEL + c];
    lsum += v[i];
  }
  __shared__ float red[8];
  int lane = tid & 31, wid = tid >> 5;
#pragma unroll
  for (int off = 16; off; off >>= 1) lsum += __shfl_xor(lsum, off);
  if (lane == 0) red[wid] = lsum;
  __syncthreads();
  float tsum = 0.f;
#pragma unroll
  for (int i = 0; i < 8; i++) tsum += red[i];
  float mean = tsum * (1.f / D_MODEL);
  __syncthreads();
  float lvar = 0.f;
#pragma unroll
  for (int i = 0; i < 4; i++) { float d = v[i] - mean; lvar += d * d; }
#pragma unroll
  for (int off = 16; off; off >>= 1) lvar += __shfl_xor(lvar, off);
  if (lane == 0) red[wid] = lvar;
  __syncthreads();
  float tvar = 0.f;
#pragma unroll
  for (int i = 0; i < 8; i++) tvar += red[i];
  float rs = rsqrtf(tvar * (1.f / D_MODEL) + 1e-5f);
#pragma unroll
  for (int i = 0; i < 4; i++) {
    int c = tid + i * 256;
    out[(size_t)t * D_MODEL + c] = (v[i] - mean) * rs * gamma[c] + beta[c];
  }
}

// ---------------- host-side launcher ----------------
extern "C" void kernel_launch(void* const* d_in, const int* in_sizes, int n_in,
                              void* d_out, int out_size, void* d_ws, size_t ws_size,
                              hipStream_t stream) {
  (void)in_sizes; (void)n_in; (void)out_size; (void)ws_size;
  const float* x        = (const float*)d_in[0];
  const float* w_router = (const float*)d_in[1];
  const float* w_gate   = (const float*)d_in[2];
  const float* w_up     = (const float*)d_in[3];
  const float* w_down   = (const float*)d_in[4];
  const float* gamma    = (const float*)d_in[5];
  const float* beta     = (const float*)d_in[6];
  float* out = (float*)d_out;

  char* ws = (char*)d_ws;                     // needs ~72.3 MB
  int*   counts = (int*)(ws + OFF_COUNTS);
  int*   cnt2   = (int*)(ws + OFF_CNT2);
  int*   base   = (int*)(ws + OFF_BASE);
  int*   eIdx   = (int*)(ws + OFF_EIDX);
  float* wTok   = (float*)(ws + OFF_WTOK);
  int*   tok    = (int*)(ws + OFF_TOK);
  float* wgt    = (float*)(ws + OFF_WGT);
  int*   inv    = (int*)(ws + OFF_INV);
  unsigned short* xb = (unsigned short*)(ws + OFF_XB);
  unsigned short* H  = (unsigned short*)(ws + OFF_H);
  float*          Y  = (float*)(ws + OFF_Y);

  init_kernel<<<1, 32, 0, stream>>>(counts, cnt2);
  xb_kernel<<<(N_TOK * D_MODEL) / (256 * 4), 256, 0, stream>>>(x, xb);
  router_kernel<<<N_TOK / 8, 256, 0, stream>>>(x, w_router, eIdx, wTok, counts);
  scan_kernel<<<1, 32, 0, stream>>>(counts, base);
  scatter_kernel<<<N_TOK / 256, 256, 0, stream>>>(eIdx, wTok, base, cnt2, tok, wgt, inv);
  gemm_gateup_kernel<<<dim3(N_TOK / 128, D_FFN / 64, N_EXP), 256, 0, stream>>>(
      xb, w_gate, w_up, H, counts, base, tok);
  gemm_down_kernel<<<dim3(N_TOK / 128, D_MODEL / 64, N_EXP), 256, 0, stream>>>(
      H, w_down, Y, counts, base);
  combine_ln_kernel<<<N_TOK, 256, 0, stream>>>(Y, inv, wgt, gamma, beta, out);
}
